// AttentionHead_6957847019746
// MI455X (gfx1250) — matrix operand order repaired and checked by reading, not compiled, not run on hardware
//
#include <hip/hip_runtime.h>

// Problem constants (match reference)
#define B_ 8
#define T_ 2048
#define C_ 1024
#define H_ 128

typedef __bf16        v16bf __attribute__((ext_vector_type(16)));
typedef float         v8f   __attribute__((ext_vector_type(8)));
typedef float         v4f   __attribute__((ext_vector_type(4)));
typedef unsigned int  v8u   __attribute__((ext_vector_type(8)));
typedef unsigned int  v4u   __attribute__((ext_vector_type(4)));

union AFrag { v16bf bf; v4u q[2]; unsigned short s[16]; };
union BFrag { v16bf bf; v8u u; };
union Pack8 { unsigned short s[8]; v4u v; };

// fp32 -> bf16, round-to-nearest-even
static __device__ inline unsigned short f2bf(float f) {
  unsigned u = __builtin_bit_cast(unsigned, f);
  unsigned r = u + 0x7FFFu + ((u >> 16) & 1u);
  return (unsigned short)(r >> 16);
}

static __device__ inline v8f wmma_bf16(v16bf a, v16bf b, v8f c) {
  // D = A(16x32 bf16) * B(32x16 bf16) + C(16x16 f32)
  return __builtin_amdgcn_wmma_f32_16x16x32_bf16(false, a, false, b, (short)0, c, false, false);
}

// gfx1250 async DMA: 16B global -> LDS, tracked by ASYNCcnt.
// lds_off = byte offset into LDS (low 32 bits of a generic __shared__ pointer).
static __device__ inline void async_ld16(unsigned lds_off, const void* g) {
  asm volatile("global_load_async_to_lds_b128 %0, %1, off"
               :: "v"(lds_off), "v"((unsigned long long)(size_t)g)
               : "memory");
}
#define WAIT_ASYNC(n) asm volatile("s_wait_asynccnt " #n ::: "memory")

// ---------------------------------------------------------------------------
// Kernel 1: x fp32 -> bf16 (8 elements / thread, vector loads+stores)
// ---------------------------------------------------------------------------
__global__ __launch_bounds__(256) void convert_x_kernel(const float* __restrict__ x,
                                                        unsigned short* __restrict__ xb,
                                                        int n8) {
  int i = blockIdx.x * blockDim.x + threadIdx.x;
  if (i >= n8) return;
  const v4f* src = (const v4f*)(x + (size_t)i * 8);
  v4f a = src[0], b = src[1];
  Pack8 p;
#pragma unroll
  for (int j = 0; j < 4; ++j) p.s[j]     = f2bf(a[j]);
#pragma unroll
  for (int j = 0; j < 4; ++j) p.s[4 + j] = f2bf(b[j]);
  *(v4u*)(xb + (size_t)i * 8) = p.v;
}

// ---------------------------------------------------------------------------
// Kernel 2: Wq/Wk/Wv [C,H] fp32 -> Wt[3][H][C] bf16 (transposed)
// ---------------------------------------------------------------------------
__global__ __launch_bounds__(256) void convert_w_kernel(const float* __restrict__ Wq,
                                                        const float* __restrict__ Wk,
                                                        const float* __restrict__ Wv,
                                                        unsigned short* __restrict__ Wt) {
  int i = blockIdx.x * blockDim.x + threadIdx.x;
  const int per = C_ * H_;
  if (i >= 3 * per) return;
  int sel = i / per, r = i % per;
  int c = r / H_, h = r % H_;
  const float* W = (sel == 0) ? Wq : ((sel == 1) ? Wk : Wv);
  Wt[(size_t)sel * per + (size_t)h * C_ + c] = f2bf(W[r]);
}

// ---------------------------------------------------------------------------
// Kernel 3: projections. grid = (B*T/64, 3), 4 waves/WG, each wave 16 t-rows.
// Wt K-tile [128 h][32 k] is async-DMA'd to LDS (double buffered) and shared
// by all 4 waves. sel 0 -> q[B,T,H], 1 -> k[B,T,H], 2 -> vT[B,H,T] (bf16).
// ---------------------------------------------------------------------------
#define WSTR 80   // bytes per h-row in LDS W tile (64 data + 16 pad, 16B mult)

__global__ __launch_bounds__(128) void proj_kernel(const unsigned short* __restrict__ xb,
                                                   const unsigned short* __restrict__ Wt,
                                                   const float* __restrict__ bq,
                                                   const float* __restrict__ bk,
                                                   const float* __restrict__ bv,
                                                   unsigned short* __restrict__ qo,
                                                   unsigned short* __restrict__ ko,
                                                   unsigned short* __restrict__ vT) {
  __shared__ __align__(16) unsigned char wbuf[2][128 * WSTR];   // 2 x 10 KB

  const int tid  = threadIdx.x;
  const int lane = tid & 31, w = tid >> 5;
  const int hi = lane >> 4, lo = lane & 15;
  const int sel = blockIdx.y;
  const int b   = blockIdx.x / (T_ / 64);
  const int tb  = (blockIdx.x % (T_ / 64)) * 64 + w * 16;

  const unsigned char* Wsel = (const unsigned char*)(Wt + (size_t)sel * C_ * H_);
  const float* bias = (sel == 0) ? bq : ((sel == 1) ? bk : bv);

  v8f zero = {0.f, 0.f, 0.f, 0.f, 0.f, 0.f, 0.f, 0.f};
  v8f acc[8];
#pragma unroll
  for (int n = 0; n < 8; ++n) acc[n] = zero;

  const unsigned short* xrow = xb + ((size_t)b * T_ + tb + lo) * C_;

  // Issue async DMA of W K-tile kb into wbuf[buf]: 512 x 16B chunks, 4/thread.
  auto issue_w = [&](int kb, int buf) {
    unsigned base = (unsigned)(size_t)&wbuf[buf][0];
#pragma unroll
    for (int j = 0; j < 4; ++j) {
      int c = tid + j * 128;            // 0..511
      int h = c >> 2, cc = c & 3;       // row h, 16B chunk within 64B row
      async_ld16(base + h * WSTR + cc * 16,
                 Wsel + (size_t)h * (C_ * 2) + (size_t)kb * 64 + cc * 16);
    }
  };

  issue_w(0, 0);
  int cur = 0;
  for (int kb = 0; kb < C_ / 32; ++kb) {
    if (kb + 1 < C_ / 32) { issue_w(kb + 1, cur ^ 1); WAIT_ASYNC(4); }
    else                  { WAIT_ASYNC(0); }
    __syncthreads();

    AFrag a;
    a.q[0] = *(const v4u*)(xrow + kb * 32 + hi * 8);
    a.q[1] = *(const v4u*)(xrow + kb * 32 + 16 + hi * 8);

    const unsigned char* wl = wbuf[cur];
#pragma unroll
    for (int n = 0; n < 8; ++n) {
      BFrag bf;
      ((v4u*)&bf.u)[0] = *(const v4u*)(wl + (n * 16 + lo) * WSTR + hi * 16);
      ((v4u*)&bf.u)[1] = *(const v4u*)(wl + (n * 16 + lo) * WSTR + 32 + hi * 16);
      acc[n] = wmma_bf16(a.bf, bf.bf, acc[n]);
    }
    __syncthreads();
    cur ^= 1;
  }

  // bias add + store bf16 (C layout: row = i + 8*hi, col = n*16 + lo)
#pragma unroll
  for (int n = 0; n < 8; ++n) {
    int h = n * 16 + lo;
    float bval = bias[h];
#pragma unroll
    for (int i = 0; i < 8; ++i) {
      int t = tb + i + 8 * hi;
      unsigned short bh = f2bf(acc[n][i] + bval);
      if (sel == 2)       vT[((size_t)b * H_ + h) * T_ + t] = bh;
      else if (sel == 1)  ko[((size_t)b * T_ + t) * H_ + h] = bh;
      else                qo[((size_t)b * T_ + t) * H_ + h] = bh;
    }
  }
}

// ---------------------------------------------------------------------------
// Kernel 4: causal flash attention. 4 waves/WG own adjacent 16-row q tiles;
// k/vT s-tiles async-DMA'd to LDS (double buffered) and shared by all waves.
// Online softmax per row-pair of lanes. EXEC stays full for all WMMA.
// ---------------------------------------------------------------------------
#define KSTR 272  // bytes per s-row of k tile (256 data + 16 pad)
#define VSTR 80   // bytes per h-row of vT tile (64 data + 16 pad)
#define SSTR 40   // floats per row of score staging tile

__global__ __launch_bounds__(128) void attn_kernel(const unsigned short* __restrict__ q,
                                                   const unsigned short* __restrict__ k,
                                                   const unsigned short* __restrict__ vT,
                                                   float* __restrict__ out) {
  __shared__ __align__(16) unsigned char kbuf[2][32 * KSTR];    // 2 x 8.5 KB
  __shared__ __align__(16) unsigned char vbuf[2][128 * VSTR];   // 2 x 10 KB
  __shared__ __align__(16) float stage[4][16 * SSTR];           // 10 KB

  const int tid  = threadIdx.x;
  const int lane = tid & 31, w = tid >> 5;
  const int hi = lane >> 4, lo = lane & 15;
  const int b   = blockIdx.x >> 5;                  // 32 q-chunks of 64 / batch
  const int qwg = (blockIdx.x & 31) * 64;           // WG's first q row
  const int qb  = qwg + w * 16;                     // wave's first q row
  float* lds = stage[w];

  // Preload q A-fragments for all 4 K-chunks of H=128
  AFrag aq[4];
  const unsigned short* qrow = q + ((size_t)b * T_ + qb + lo) * H_;
#pragma unroll
  for (int hb = 0; hb < 4; ++hb) {
    aq[hb].q[0] = *(const v4u*)(qrow + hb * 32 + hi * 8);
    aq[hb].q[1] = *(const v4u*)(qrow + hb * 32 + 16 + hi * 8);
  }

  v8f zero = {0.f, 0.f, 0.f, 0.f, 0.f, 0.f, 0.f, 0.f};
  v8f acco[8];
#pragma unroll
  for (int n = 0; n < 8; ++n) acco[n] = zero;

  float m = -3.0e38f;   // running row max (lane pair l / l+16)
  float lsum = 0.f;     // running row sum

  // Issue async DMA of k tile [32 s][128 h] and vT tile [128 h][32 s]
  // for s-block blk into buffers: 8 x 16B chunks per thread.
  auto issue_blk = [&](int blk, int buf) {
    const int sblk = blk * 32;
    const unsigned char* kg = (const unsigned char*)(k + ((size_t)b * T_ + sblk) * H_);
    const unsigned char* vg = (const unsigned char*)(vT + (size_t)b * H_ * T_ + sblk);
    unsigned kb_off = (unsigned)(size_t)&kbuf[buf][0];
    unsigned vb_off = (unsigned)(size_t)&vbuf[buf][0];
#pragma unroll
    for (int j = 0; j < 4; ++j) {
      int c = tid + j * 128;                 // 0..511
      int kr = c >> 4, kc = c & 15;          // k: 32 rows x 16 chunks
      async_ld16(kb_off + kr * KSTR + kc * 16,
                 kg + (size_t)kr * (H_ * 2) + kc * 16);
      int vr = c >> 2, vc = c & 3;           // vT: 128 rows x 4 chunks
      async_ld16(vb_off + vr * VSTR + vc * 16,
                 vg + (size_t)vr * (T_ * 2) + vc * 16);
    }
  };

  const int nblk = (qwg >> 5) + 2;           // covers s <= qwg + 63
  issue_blk(0, 0);
  int cur = 0;

  for (int blk = 0; blk < nblk; ++blk) {
    const int sblk = blk * 32;
    if (blk + 1 < nblk) { issue_blk(blk + 1, cur ^ 1); WAIT_ASYNC(8); }
    else                { WAIT_ASYNC(0); }
    __syncthreads();

    if (sblk <= qb + 15) {   // wave-uniform causal skip
      const unsigned char* kl = kbuf[cur];
      const unsigned char* vl = vbuf[cur];

      // ---- S = q @ k^T for 16x32 block (two 16x16 WMMA tiles) ----
      v8f s0 = zero, s1 = zero;
#pragma unroll
      for (int hb = 0; hb < 4; ++hb) {
        BFrag bk0, bk1;
        ((v4u*)&bk0.u)[0] = *(const v4u*)(kl + lo * KSTR        + hb * 64 + hi * 16);
        ((v4u*)&bk0.u)[1] = *(const v4u*)(kl + lo * KSTR        + hb * 64 + 32 + hi * 16);
        ((v4u*)&bk1.u)[0] = *(const v4u*)(kl + (16 + lo) * KSTR + hb * 64 + hi * 16);
        ((v4u*)&bk1.u)[1] = *(const v4u*)(kl + (16 + lo) * KSTR + hb * 64 + 32 + hi * 16);
        s0 = wmma_bf16(aq[hb].bf, bk0.bf, s0);
        s1 = wmma_bf16(aq[hb].bf, bk1.bf, s1);
      }

      // ---- stage C-layout scores to per-wave LDS [row][col] fp32 ----
#pragma unroll
      for (int i = 0; i < 8; ++i) {
        int r = i + 8 * hi;
        lds[r * SSTR + lo]      = s0[i];
        lds[r * SSTR + 16 + lo] = s1[i];
      }
      __builtin_amdgcn_wave_barrier();
      asm volatile("s_wait_dscnt 0" ::: "memory");

      // ---- reload in A-fragment ordering: row = lo, K chunks per half ----
      const int r = lo;
      const int ka = hi * 8;
      float p[16];
      v4f t0 = *(const v4f*)&lds[r * SSTR + ka];
      v4f t1 = *(const v4f*)&lds[r * SSTR + ka + 4];
      v4f t2 = *(const v4f*)&lds[r * SSTR + 16 + ka];
      v4f t3 = *(const v4f*)&lds[r * SSTR + 16 + ka + 4];
#pragma unroll
      for (int j = 0; j < 4; ++j) { p[j] = t0[j]; p[4+j] = t1[j]; p[8+j] = t2[j]; p[12+j] = t3[j]; }

      // ---- causal mask (diagonal-touching blocks only) ----
      if (sblk + 31 > qb) {
        int qr = qb + r;
#pragma unroll
        for (int j = 0; j < 8; ++j) {
          if (sblk + ka + j      > qr) p[j]     = -3.0e38f;
          if (sblk + 16 + ka + j > qr) p[8 + j] = -3.0e38f;
        }
      }

      // ---- online softmax (row split across lane pair l / l+16) ----
      float bm = p[0];
#pragma unroll
      for (int j = 1; j < 16; ++j) bm = fmaxf(bm, p[j]);
      bm = fmaxf(bm, __shfl_xor(bm, 16, 32));
      float mnew  = fmaxf(m, bm);
      float scale = __expf(m - mnew);

      AFrag ap;
      float bs = 0.f;
#pragma unroll
      for (int j = 0; j < 16; ++j) {
        float e = __expf(p[j] - mnew);
        bs += e;
        ap.s[j] = f2bf(e);
      }
      bs += __shfl_xor(bs, 16, 32);
      lsum = lsum * scale + bs;
      m = mnew;

      // ---- rescale output accumulators (per-row factor -> C layout) ----
      float scr[8];
#pragma unroll
      for (int i = 0; i < 8; ++i) scr[i] = __shfl(scale, i + 8 * hi, 32);
#pragma unroll
      for (int n = 0; n < 8; ++n)
#pragma unroll
        for (int i = 0; i < 8; ++i) acco[n][i] *= scr[i];

      // ---- O += P @ V (vT tile from LDS) ----
#pragma unroll
      for (int n = 0; n < 8; ++n) {
        BFrag bv;
        ((v4u*)&bv.u)[0] = *(const v4u*)(vl + (n * 16 + lo) * VSTR + hi * 16);
        ((v4u*)&bv.u)[1] = *(const v4u*)(vl + (n * 16 + lo) * VSTR + 32 + hi * 16);
        acco[n] = wmma_bf16(ap.bf, bv.bf, acco[n]);
      }
    }

    __syncthreads();
    cur ^= 1;
  }

  // ---- epilogue: divide by row sums, store fp32 ----
  float linv[8];
#pragma unroll
  for (int i = 0; i < 8; ++i) linv[i] = 1.f / __shfl(lsum, i + 8 * hi, 32);
#pragma unroll
  for (int n = 0; n < 8; ++n) {
    int h = n * 16 + lo;
#pragma unroll
    for (int i = 0; i < 8; ++i) {
      int t = qb + i + 8 * hi;
      out[((size_t)b * T_ + t) * H_ + h] = acco[n][i] * linv[i];
    }
  }
}

// ---------------------------------------------------------------------------
// Launch
// ---------------------------------------------------------------------------
extern "C" void kernel_launch(void* const* d_in, const int* in_sizes, int n_in,
                              void* d_out, int out_size, void* d_ws, size_t ws_size,
                              hipStream_t stream) {
  const float* x  = (const float*)d_in[0];
  const float* Wq = (const float*)d_in[1];
  const float* bq = (const float*)d_in[2];
  const float* Wk = (const float*)d_in[3];
  const float* bk = (const float*)d_in[4];
  const float* Wv = (const float*)d_in[5];
  const float* bv = (const float*)d_in[6];
  float* out = (float*)d_out;

  // Workspace layout (bytes)
  char* ws = (char*)d_ws;
  const size_t XB_OFF = 0;                                  // bf16 x     [B,T,C]
  const size_t WT_OFF = XB_OFF + (size_t)B_ * T_ * C_ * 2;  // bf16 Wt [3][H,C]
  const size_t Q_OFF  = WT_OFF + (size_t)3 * C_ * H_ * 2;   // bf16 q     [B,T,H]
  const size_t K_OFF  = Q_OFF  + (size_t)B_ * T_ * H_ * 2;  // bf16 k     [B,T,H]
  const size_t VT_OFF = K_OFF  + (size_t)B_ * T_ * H_ * 2;  // bf16 vT    [B,H,T]

  unsigned short* xb = (unsigned short*)(ws + XB_OFF);
  unsigned short* Wt = (unsigned short*)(ws + WT_OFF);
  unsigned short* dq = (unsigned short*)(ws + Q_OFF);
  unsigned short* dk = (unsigned short*)(ws + K_OFF);
  unsigned short* vT = (unsigned short*)(ws + VT_OFF);

  const int n8 = B_ * T_ * C_ / 8;
  convert_x_kernel<<<(n8 + 255) / 256, 256, 0, stream>>>(x, xb, n8);

  const int nw = 3 * C_ * H_;
  convert_w_kernel<<<(nw + 255) / 256, 256, 0, stream>>>(Wq, Wk, Wv, Wt);

  dim3 pg(B_ * T_ / 64, 3);
  proj_kernel<<<pg, 128, 0, stream>>>(xb, Wt, bq, bk, bv, dq, dk, vT);

  attn_kernel<<<B_ * T_ / 64, 128, 0, stream>>>(dq, dk, vT, out);
}
